// FineGrainedOpLstmCellV1_61486751809783
// MI455X (gfx1250) — matrix operand order repaired and checked by reading, not compiled, not run on hardware
//
#include <hip/hip_runtime.h>

// ---- CDNA5 WMMA / TDM vector types (probe-confirmed signatures) ----
typedef __attribute__((ext_vector_type(16))) __bf16       v16bf;
typedef __attribute__((ext_vector_type(2)))  __bf16       v2bf;
typedef __attribute__((ext_vector_type(8)))  float        v8f;
typedef __attribute__((ext_vector_type(4)))  unsigned int u32x4;
typedef __attribute__((ext_vector_type(8)))  int          i32x8;
typedef __attribute__((ext_vector_type(4)))  int          i32x4;

#define HDIM  1024   // hidden == input size
#define BATCH 4096
#define BM    128    // rows (batch) per workgroup
#define BN    32     // cols (hidden) per workgroup
#define KC    32     // K chunk per iteration (one wmma K)
#define LDA   40     // padded LDS row length in ushorts (80 B, 16B-aligned)

struct alignas(16) F4 { float f[4]; };

// pack two fp32 -> two bf16 in one dword: a -> [15:0], b -> [31:16]
#if __has_builtin(__builtin_amdgcn_cvt_pk_bf16_f32)
__device__ __forceinline__ unsigned pk2bf(float a, float b) {
  union { v2bf v; unsigned u; } cv;
  cv.v = __builtin_amdgcn_cvt_pk_bf16_f32(a, b);   // single v_cvt_pk_bf16_f32
  return cv.u;
}
#else
// Builtin not declared on this toolchain (round-4 histogram proved the manual
// path compiled) -> probe the ISA directly: one VALU op per bf16 pair.
__device__ __forceinline__ unsigned pk2bf(float a, float b) {
  unsigned r;
  asm("v_cvt_pk_bf16_f32 %0, %1, %2" : "=v"(r) : "v"(a), "v"(b));
  return r;
}
#endif

__device__ __forceinline__ float fast_sigmoid(float x) {
  return 1.0f / (1.0f + __expf(-x));
}
__device__ __forceinline__ float fast_tanh(float x) {
  const float e = __expf(-2.0f * x);
  return (1.0f - e) / (1.0f + e);
}

// ---- Tensor Data Mover: async copy of a 32-float row (2D tile 32x1, fp32) to LDS ----
// D# built per CDNA5 ISA (group0/group1 bitfields), 6-arg
// (clang-23 / therock) builtin form: (g0, g1, g2, g3, g4, cpol).
__device__ __forceinline__ void tdm_load_row32(const float* gsrc, unsigned lds_off) {
  const unsigned long long ga = (unsigned long long)(uintptr_t)gsrc;
  u32x4 g0;
  g0[0] = 1u;                                   // count=1, user mode
  g0[1] = lds_off;                              // lds_addr (bytes)
  g0[2] = (unsigned)(ga & 0xFFFFFFFFull);       // global_addr[31:0]
  g0[3] = (unsigned)((ga >> 32) & 0x01FFFFFFull) | (2u << 30);  // addr[56:32] | type=2
  i32x8 g1;
  g1[0] = (2 << 16);     // workgroup_mask=0, data_size=2 (4 bytes)
  g1[1] = (32 << 16);    // tensor_dim0 = 32 (low 16 bits in word1[31:16])
  g1[2] = (1 << 16);     // tensor_dim0 hi=0 | tensor_dim1 = 1
  g1[3] = (32 << 16);    // tensor_dim1 hi=0 | tile_dim0 = 32
  g1[4] = 1;             // tile_dim1 = 1, tile_dim2 = 0
  g1[5] = 32;            // tensor_dim0_stride = 32
  g1[6] = 0;
  g1[7] = 0;
  const i32x4 gz4 = {0, 0, 0, 0};               // groups 2/3 unused (2D tensor)
  const i32x8 gz8 = {0, 0, 0, 0, 0, 0, 0, 0};   // trailing group (6-arg form)
  __builtin_amdgcn_tensor_load_to_lds(g0, g1, gz4, gz4, gz8, 0);
}

__global__ __launch_bounds__(256)
void lstm_cell_wmma_bf16(const float* __restrict__ x,
                         const float* __restrict__ h_prev,
                         const float* __restrict__ c_prev,
                         const float* __restrict__ wxi, const float* __restrict__ wxf,
                         const float* __restrict__ wxo, const float* __restrict__ wxc,
                         const float* __restrict__ whi, const float* __restrict__ whf,
                         const float* __restrict__ who, const float* __restrict__ whc,
                         const float* __restrict__ bi,  const float* __restrict__ bfv,
                         const float* __restrict__ bo,  const float* __restrict__ bcv,
                         float* __restrict__ out)
{
  __shared__ __align__(16) unsigned short As[BM][LDA];     // A tile, bf16, row-major
  __shared__ __align__(16) unsigned short Bs[4][BN][LDA];  // B tiles, bf16, TRANSPOSED [gate][n][k]
  __shared__ __align__(16) float BiasLds[4][BN];           // per-gate bias slice (TDM target)

  const int t    = threadIdx.x;
  const int lane = t & 31;
  const int wave = t >> 5;      // 8 waves
  const int wr   = wave >> 1;   // 0..3 : 32-row band
  const int wc   = wave & 1;    // 0..1 : 16-col band
  const int m0   = blockIdx.y * BM;
  const int n0   = blockIdx.x * BN;

  // --- kick off async TDM copies of this tile's bias slices (fp32, raw copy) ---
  tdm_load_row32(bi  + n0, (unsigned)(uintptr_t)&BiasLds[0][0]);
  tdm_load_row32(bfv + n0, (unsigned)(uintptr_t)&BiasLds[1][0]);
  tdm_load_row32(bo  + n0, (unsigned)(uintptr_t)&BiasLds[2][0]);
  tdm_load_row32(bcv + n0, (unsigned)(uintptr_t)&BiasLds[3][0]);

  // cooperative-loader indices
  const int a_tr = t >> 3;          // 0..31 (row within A tile per pass)
  const int a_tc = (t & 7) * 4;     // col (k) within chunk
  const int g_ld = t >> 6;          // gate this thread stages (64 thr/gate)
  const int uu   = t & 63;
  const int b_kb = (uu >> 3) * 4;   // k block within chunk (0..28)
  const int b_nb = (uu & 7) * 4;    // n block within tile  (0..28)

  // weight pointers for this thread's gate (no dynamically-indexed local array -> no scratch)
  const float* WgX = (g_ld == 0) ? wxi : (g_ld == 1) ? wxf : (g_ld == 2) ? wxo : wxc;
  const float* WgH = (g_ld == 0) ? whi : (g_ld == 1) ? whf : (g_ld == 2) ? who : whc;

  v8f acc[4][2];
#pragma unroll
  for (int g = 0; g < 4; ++g)
#pragma unroll
    for (int mi = 0; mi < 2; ++mi)
      acc[g][mi] = (v8f){0.f, 0.f, 0.f, 0.f, 0.f, 0.f, 0.f, 0.f};

  // fragment indices (contiguous k order: lane half h holds k = 16h + s)
  const int fcol = wc * 16 + (lane & 15);      // B col
  const int kb   = (lane < 16) ? 0 : 16;       // B k-group
  const int ka   = (lane < 16) ? 0 : 16;       // A k-group

#pragma unroll
  for (int half = 0; half < 2; ++half) {       // half 0: x@Wx, half 1: h@Wh
    const float* Asrc = half ? h_prev : x;
    const float* Wsrc = half ? WgH : WgX;

    for (int kc = 0; kc < HDIM; kc += KC) {
      // --- stage A tile: 128x32 fp32 -> bf16 LDS (coalesced float4, packed b64 stores) ---
#pragma unroll
      for (int p = 0; p < 4; ++p) {
        const int r = a_tr + p * 32;
        const F4 v = *reinterpret_cast<const F4*>(&Asrc[(size_t)(m0 + r) * HDIM + kc + a_tc]);
        *reinterpret_cast<uint2*>(&As[r][a_tc]) =
            make_uint2(pk2bf(v.f[0], v.f[1]), pk2bf(v.f[2], v.f[3]));
      }
      // --- stage B tile for this thread's gate: 4x4 register-block transpose ---
      {
        const float* wp = &Wsrc[(size_t)(kc + b_kb) * HDIM + n0 + b_nb];
        const F4 w0 = *reinterpret_cast<const F4*>(wp);
        const F4 w1 = *reinterpret_cast<const F4*>(wp + HDIM);
        const F4 w2 = *reinterpret_cast<const F4*>(wp + 2 * HDIM);
        const F4 w3 = *reinterpret_cast<const F4*>(wp + 3 * HDIM);
#pragma unroll
        for (int i = 0; i < 4; ++i) {
          *reinterpret_cast<uint2*>(&Bs[g_ld][b_nb + i][b_kb]) =
              make_uint2(pk2bf(w0.f[i], w1.f[i]), pk2bf(w2.f[i], w3.f[i]));
        }
      }

      // prefetch next A chunk (global_prefetch_b8)
      if (kc + KC < HDIM)
        __builtin_prefetch(&Asrc[(size_t)(m0 + a_tr) * HDIM + kc + KC + a_tc], 0, 0);

      __syncthreads();

      // --- fragments + WMMA ---
      union Frag { uint4 q[2]; v16bf v; };
      Frag bfrag[4], afrag;
#pragma unroll
      for (int g = 0; g < 4; ++g) {
        bfrag[g].q[0] = *reinterpret_cast<const uint4*>(&Bs[g][fcol][kb]);
        bfrag[g].q[1] = *reinterpret_cast<const uint4*>(&Bs[g][fcol][kb + 8]);
      }
#pragma unroll
      for (int mi = 0; mi < 2; ++mi) {
        const int arow = wr * 32 + mi * 16 + (lane & 15);
        afrag.q[0] = *reinterpret_cast<const uint4*>(&As[arow][ka]);
        afrag.q[1] = *reinterpret_cast<const uint4*>(&As[arow][ka + 8]);
#pragma unroll
        for (int g = 0; g < 4; ++g) {
          acc[g][mi] = __builtin_amdgcn_wmma_f32_16x16x32_bf16(
              false, afrag.v, false, bfrag[g].v,
              (short)0, acc[g][mi], false, false);
        }
      }
      __syncthreads();
    }
  }

  // --- fused LSTM epilogue (biases arrived via TDM long ago; wait on TENSORcnt) ---
  __builtin_amdgcn_s_wait_tensorcnt(0);

  const int nn    = wc * 16 + (lane & 15);
  const int colH  = n0 + nn;
  const int rbase = m0 + wr * 32 + ((lane < 16) ? 0 : 8);
  const float vbi = BiasLds[0][nn];
  const float vbf = BiasLds[1][nn];
  const float vbo = BiasLds[2][nn];
  const float vbc = BiasLds[3][nn];

#pragma unroll
  for (int mi = 0; mi < 2; ++mi) {
#pragma unroll
    for (int j = 0; j < 8; ++j) {
      const int row = rbase + mi * 16 + j;
      const float gi = fast_sigmoid(acc[0][mi][j] + vbi);
      const float gf = fast_sigmoid(acc[1][mi][j] + vbf);
      const float go = fast_sigmoid(acc[2][mi][j] + vbo);
      const float gc = fast_tanh   (acc[3][mi][j] + vbc);
      const float cp = c_prev[(size_t)row * HDIM + colH];
      const float cN = gf * cp + gi * gc;
      const float hN = go * fast_tanh(cN);
      out[(size_t)row * HDIM + colH] = hN;                          // h (output 0)
      out[(size_t)BATCH * HDIM + (size_t)row * HDIM + colH] = cN;   // c (output 1)
    }
  }
}

extern "C" void kernel_launch(void* const* d_in, const int* in_sizes, int n_in,
                              void* d_out, int out_size, void* d_ws, size_t ws_size,
                              hipStream_t stream) {
  (void)in_sizes; (void)n_in; (void)out_size; (void)d_ws; (void)ws_size;
  // setup_inputs order: x, h_prev, c_prev, igx, igu, ib, fgx, fgu, fb,
  //                     ogx, ogu, ob, cgx, cgu, cb
  const float* x      = (const float*)d_in[0];
  const float* h_prev = (const float*)d_in[1];
  const float* c_prev = (const float*)d_in[2];
  const float* igx = (const float*)d_in[3];
  const float* igu = (const float*)d_in[4];
  const float* ib  = (const float*)d_in[5];
  const float* fgx = (const float*)d_in[6];
  const float* fgu = (const float*)d_in[7];
  const float* fb  = (const float*)d_in[8];
  const float* ogx = (const float*)d_in[9];
  const float* ogu = (const float*)d_in[10];
  const float* ob  = (const float*)d_in[11];
  const float* cgx = (const float*)d_in[12];
  const float* cgu = (const float*)d_in[13];
  const float* cb  = (const float*)d_in[14];
  float* out = (float*)d_out;

  dim3 grid(HDIM / BN, BATCH / BM);   // (32, 32) workgroups
  dim3 block(256);                    // 8 wave32 per workgroup
  lstm_cell_wmma_bf16<<<grid, block, 0, stream>>>(
      x, h_prev, c_prev,
      igx, fgx, ogx, cgx,
      igu, fgu, ogu, cgu,
      ib, fb, ob, cb,
      out);
}
